// PolicyHead_77644418777369
// MI455X (gfx1250) — matrix operand order repaired and checked by reading, not compiled, not run on hardware
//
#include <hip/hip_runtime.h>
#include <math.h>

// ---------------------------------------------------------------------------
// Problem constants (fixed by the reference)
// ---------------------------------------------------------------------------
#define BATCH   32
#define SEQ     1024
#define TSEQ    2048
#define DMODEL  256
#define BSROWS  (BATCH * SEQ)    // 32768
#define BTROWS  (BATCH * TSEQ)   // 65536

// ---------------------------------------------------------------------------
// CDNA5 WMMA helpers (bf16 inputs, fp32 accumulate)
// ---------------------------------------------------------------------------
typedef __attribute__((ext_vector_type(16))) __bf16 bf16x16;
typedef __attribute__((ext_vector_type(8)))  float  f32x8;
typedef __attribute__((ext_vector_type(2)))  float  f32x2;
typedef __attribute__((ext_vector_type(2)))  __bf16 bf16x2;

union Frag {
    bf16x16 v;
    uint4   u[2];
};

static __device__ __forceinline__ f32x8 fzero8() {
    f32x8 z = {0.f, 0.f, 0.f, 0.f, 0.f, 0.f, 0.f, 0.f};
    return z;
}

static __device__ __forceinline__ f32x8 wmma_bf16(bf16x16 a, bf16x16 b, f32x8 c) {
    // args: (neg_a, A, neg_b, B, c_mod, C, reuse_a, reuse_b)
    return __builtin_amdgcn_wmma_f32_16x16x32_bf16(false, a, false, b, (short)0, c,
                                                   false, false);
}

// Hardware f32->bf16 conversions (lowers to v_cvt_pk_bf16_f32 on gfx1250).
static __device__ __forceinline__ unsigned pack2bf(float a, float b) {
    union { bf16x2 h; unsigned u; } cv;
    f32x2 f = {a, b};
    cv.h = __builtin_convertvector(f, bf16x2);
    return cv.u;
}
static __device__ __forceinline__ unsigned short f2bf(float x) {
    union { __bf16 h; unsigned short u; } cv;
    cv.h = (__bf16)x;
    return cv.u;
}

// ---------------------------------------------------------------------------
// CDNA5 async global->LDS copy (16 bytes, per-lane addresses; ASYNCcnt).
// The low 32 bits of a flat shared-memory pointer are the wave-relative LDS
// byte offset (aperture scheme, ISA 10.2), which is exactly what the VDST
// operand of GLOBAL_LOAD_ASYNC_TO_LDS_* expects.
// ---------------------------------------------------------------------------
static __device__ __forceinline__ void async_copy_b128(void* lds_dst, const void* gsrc) {
    unsigned lds_off = (unsigned)(unsigned long long)lds_dst;
    asm volatile("global_load_async_to_lds_b128 %0, %1, off"
                 :: "v"(lds_off), "v"(gsrc)
                 : "memory");
}
template <int N>
static __device__ __forceinline__ void wait_async_n() {
    asm volatile("s_wait_asynccnt %0" :: "n"(N) : "memory");
}

// ---------------------------------------------------------------------------
// Fragment address maps used by this file (contiguous k order, wave32)
//   A (16x32, bf16): lane = m + 16*(k/16),     elem = k%16
//   B (32x16, bf16): lane = n + 16*(k/16),     elem = k%16
//   C (16x16, f32) : vgpr r -> row = r + 8*(lane/16), col = lane%16
// Each fragment stored in LDS as 512 bf16: addr16 = frag*512 + lane*16 + elem
// ---------------------------------------------------------------------------

// ===========================================================================
// GEMM:  C[M,N] = act(A[M,K] @ W[K,N] + bias [+ resid])   (A,W,bias fp32)
// block = 128 threads (4 waves), tile 64x64, K stepped by 32, bf16 WMMA core.
// M % 64 == 0, N % 64 == 0, K % 8 == 0 (groups of 8 k are all-in or all-out).
// OUT16T writes bf16 transposed per batch: C16T[(batch*N + col)*seqT + t].
// ===========================================================================
template <bool RELU, bool RES, bool OUT32, bool OUT16, bool OUT16T>
__global__ __launch_bounds__(128)
void gemm_kernel(const float* __restrict__ A, const float* __restrict__ W,
                 const float* __restrict__ bias, const float* __restrict__ resid,
                 float* __restrict__ C32, unsigned short* __restrict__ C16,
                 unsigned short* __restrict__ C16T, int seqT,
                 int M, int K, int N)
{
    __shared__ __align__(16) unsigned short lA[4 * 512];   // 64x32 bf16, A-frag
    __shared__ __align__(16) unsigned short lB[4 * 512];   // 32x64 bf16, B-frag
    const int tid  = threadIdx.x;
    const int wave = tid >> 5;
    const int lane = tid & 31;
    const int m0   = blockIdx.y << 6;
    const int n0   = blockIdx.x << 6;

    f32x8 c[4];
#pragma unroll
    for (int g = 0; g < 4; ++g) c[g] = fzero8();

    for (int k0 = 0; k0 < K; k0 += 32) {
        __syncthreads();
        // ---- stage A tile (64 rows x 32 k) into A-fragment layout ----
        for (int u = tid; u < 256; u += 128) {
            int m  = u >> 2;
            int d0 = (u & 3) << 3;
            int kb = k0 + d0;
            uint4 pk;
            if (kb + 8 <= K) {   // K % 8 == 0: group fully in- or out-of-bounds
                const float4* ap =
                    reinterpret_cast<const float4*>(&A[(size_t)(m0 + m) * K + kb]);
                float4 a0 = ap[0], a1 = ap[1];
                pk.x = pack2bf(a0.x, a0.y);
                pk.y = pack2bf(a0.z, a0.w);
                pk.z = pack2bf(a1.x, a1.y);
                pk.w = pack2bf(a1.z, a1.w);
            } else {
                pk.x = pk.y = pk.z = pk.w = 0u;
            }
            int mm     = m & 15;
            int f      = m >> 4;
            int laneT  = mm + ((d0 >> 4) << 4);
            int elem0  = d0 & 15;
            *reinterpret_cast<uint4*>(&lA[f * 512 + laneT * 16 + elem0]) = pk;
        }
        // ---- stage W tile (32 k x 64 n) into B-fragment layout ----
        for (int u = tid; u < 256; u += 128) {
            int nn  = u & 63;
            int kk0 = (u >> 6) << 3;
            int kb  = k0 + kk0;
            uint4 pk;
            if (kb + 8 <= K) {
                float v[8];
#pragma unroll
                for (int j = 0; j < 8; ++j)
                    v[j] = W[(size_t)(kb + j) * N + n0 + nn];
                pk.x = pack2bf(v[0], v[1]);
                pk.y = pack2bf(v[2], v[3]);
                pk.z = pack2bf(v[4], v[5]);
                pk.w = pack2bf(v[6], v[7]);
            } else {
                pk.x = pk.y = pk.z = pk.w = 0u;
            }
            int g     = nn >> 4;
            int laneT = (nn & 15) + ((kk0 >> 4) << 4);
            int elem0 = kk0 & 15;
            *reinterpret_cast<uint4*>(&lB[g * 512 + laneT * 16 + elem0]) = pk;
        }
        __syncthreads();

        Frag a;
        a.u[0] = *reinterpret_cast<const uint4*>(&lA[wave * 512 + lane * 16]);
        a.u[1] = *reinterpret_cast<const uint4*>(&lA[wave * 512 + lane * 16 + 8]);
#pragma unroll
        for (int g = 0; g < 4; ++g) {
            Frag b;
            b.u[0] = *reinterpret_cast<const uint4*>(&lB[g * 512 + lane * 16]);
            b.u[1] = *reinterpret_cast<const uint4*>(&lB[g * 512 + lane * 16 + 8]);
            c[g] = wmma_bf16(a.v, b.v, c[g]);
        }
    }

    const int grp   = lane >> 4;
    const int nlow  = lane & 15;
    const int batch = OUT16T ? (m0 / seqT) : 0;   // tile is within one batch
#pragma unroll
    for (int g = 0; g < 4; ++g) {
        int col  = n0 + (g << 4) + nlow;
        float bv = bias[col];
#pragma unroll
        for (int r = 0; r < 8; ++r) {
            size_t row = (size_t)m0 + wave * 16 + (grp << 3) + r;
            float v = c[g][r] + bv;
            if (RES)  v += resid[row * N + col];
            if (RELU) v  = fmaxf(v, 0.f);
            if (OUT32) C32[row * N + col] = v;
            if (OUT16) C16[row * N + col] = f2bf(v);
            if (OUT16T) {
                size_t t = row - (size_t)batch * seqT;
                C16T[((size_t)batch * N + col) * seqT + t] = f2bf(v);
            }
        }
    }
}

// ===========================================================================
// Flash-style single-head attention.
//   Q, K: bf16 row-major [batch*seq, INNER]
//   Vt  : bf16 transposed [batch][d][t]  (written by the V-projection GEMM)
// All tile staging uses GLOBAL_LOAD_ASYNC_TO_LDS_B128 with the fragment
// swizzle folded into the LDS destination addresses (zero-VALU staging).
// DB=true: K/V tiles are double-buffered; the next tile's async copies are
// issued before computing on the current one, and a partial
// s_wait_asynccnt (<= ops of the in-flight prefetch group) is used so the
// HBM transfer overlaps the WMMA work (in-order async completion).
// Emits O = softmax(QK^T * scale) V and optionally the RAW softmax
// denominator sum(exp(logits)) = l * exp(m) per row.
// ===========================================================================
template <int INNER, int NW, bool WRITE_DENOM, bool DB>
__global__ __launch_bounds__(NW * 32)
void flash_kernel(const unsigned short* __restrict__ Qg,
                  const unsigned short* __restrict__ Kg,
                  const unsigned short* __restrict__ Vt,
                  float* __restrict__ Og, float* __restrict__ denom,
                  int seq_q, int seq_k, float scale)
{
    constexpr int KC   = INNER / 32;  // k-chunks for logits
    constexpr int VC   = INNER / 16;  // output column chunks
    constexpr int NT   = NW * 32;
    constexpr int NBUF = DB ? 2 : 1;
    constexpr int KOPS = (32 * INNER / 8) / NT;  // async ops per thread per K tile
    constexpr int VOPS = (4 * INNER) / NT;       // async ops per thread per V tile
    __shared__ __align__(16) unsigned short lQ[NW * KC * 512];
    __shared__ __align__(16) unsigned short lK[NBUF * KC * 2 * 512];
    __shared__ __align__(16) unsigned short lV[NBUF * VC * 512];
    __shared__ __align__(16) unsigned short lP[NW * 512];

    const int tid   = threadIdx.x;
    const int wave  = tid >> 5;
    const int lane  = tid & 31;
    const int grp   = lane >> 4;
    const int nn    = lane & 15;
    const int batch = blockIdx.y;
    const int q0    = blockIdx.x * (NW * 16);
    const size_t qrow0 = (size_t)batch * seq_q + q0;
    const size_t krow0 = (size_t)batch * seq_k;
    const size_t vbase = (size_t)batch * INNER * seq_k;

    // ---- one-time Q staging into A-fragment layout (async, swizzled dst) ----
    for (int u = tid; u < NW * 16 * INNER / 8; u += NT) {
        int m  = u / (INNER / 8);
        int d0 = (u % (INNER / 8)) * 8;
        int w = m >> 4, mm = m & 15;
        int kc = d0 >> 5, kk0 = d0 & 31;
        int laneT = mm + ((kk0 >> 4) << 4);
        int elem0 = kk0 & 15;
        async_copy_b128(&lQ[(w * KC + kc) * 512 + laneT * 16 + elem0],
                        &Qg[(qrow0 + m) * INNER + d0]);
    }

    // Stage one 32-key tile (K + V fragments) into buffer `buf`.
    auto stage_tile = [&](int t0, int buf) {
        for (int u = tid; u < 32 * INNER / 8; u += NT) {
            int t  = u / (INNER / 8);
            int d0 = (u % (INNER / 8)) * 8;
            int h = t >> 4, tt = t & 15;
            int kc = d0 >> 5, kk0 = d0 & 31;
            int laneT = tt + ((kk0 >> 4) << 4);
            int elem0 = kk0 & 15;
            async_copy_b128(
                &lK[(buf * KC * 2 + kc * 2 + h) * 512 + laneT * 16 + elem0],
                &Kg[(krow0 + t0 + t) * INNER + d0]);
        }
        // VT row d holds V[:, d] contiguously over t: 16-byte asyncs land
        // directly in the B-fragment layout.
        for (int u = tid; u < 4 * INNER; u += NT) {
            int d  = u >> 2;       // 0..INNER-1
            int s8 = u & 3;        // which 8 keys
            int vc = d >> 4;
            int laneT = (d & 15) + ((s8 >> 1) << 4);
            int elem0 = (s8 & 1) << 3;
            async_copy_b128(&lV[(buf * VC + vc) * 512 + laneT * 16 + elem0],
                            &Vt[vbase + (size_t)d * seq_k + t0 + s8 * 8]);
        }
    };

    f32x8 acc[VC];
#pragma unroll
    for (int vc = 0; vc < VC; ++vc) acc[vc] = fzero8();
    float mrow[8], lrow[8];
#pragma unroll
    for (int r = 0; r < 8; ++r) { mrow[r] = -1e30f; lrow[r] = 0.f; }

    const int ntiles = seq_k >> 5;
    if (DB) stage_tile(0, 0);

    for (int i = 0; i < ntiles; ++i) {
        const int buf = DB ? (i & 1) : 0;
        if (DB) {
            if (i + 1 < ntiles) {
                stage_tile((i + 1) << 5, buf ^ 1);   // prefetch next tile
                wait_async_n<KOPS + VOPS>();         // current tile landed
            } else {
                wait_async_n<0>();
            }
        } else {
            stage_tile(i << 5, 0);
            wait_async_n<0>();
        }
        __syncthreads();

        // ---- logits: 16 (rows) x 32 (keys) per wave ----
        const unsigned short* lKb = &lK[buf * KC * 2 * 512];
        f32x8 c0 = fzero8(), c1 = fzero8();
#pragma unroll
        for (int kc = 0; kc < KC; ++kc) {
            Frag a, b0, b1;
            a.u[0]  = *reinterpret_cast<const uint4*>(&lQ[(wave * KC + kc) * 512 + lane * 16]);
            a.u[1]  = *reinterpret_cast<const uint4*>(&lQ[(wave * KC + kc) * 512 + lane * 16 + 8]);
            b0.u[0] = *reinterpret_cast<const uint4*>(&lKb[(kc * 2) * 512 + lane * 16]);
            b0.u[1] = *reinterpret_cast<const uint4*>(&lKb[(kc * 2) * 512 + lane * 16 + 8]);
            b1.u[0] = *reinterpret_cast<const uint4*>(&lKb[(kc * 2 + 1) * 512 + lane * 16]);
            b1.u[1] = *reinterpret_cast<const uint4*>(&lKb[(kc * 2 + 1) * 512 + lane * 16 + 8]);
            c0 = wmma_bf16(a.v, b0.v, c0);
            c1 = wmma_bf16(a.v, b1.v, c1);
        }

        // ---- online softmax; write P tile to LDS in A-fragment layout ----
#pragma unroll
        for (int r = 0; r < 8; ++r) {
            float s0 = c0[r] * scale, s1 = c1[r] * scale;
            float mx = fmaxf(s0, s1);
#pragma unroll
            for (int off = 8; off >= 1; off >>= 1)
                mx = fmaxf(mx, __shfl_xor(mx, off, 16));
            float mn = fmaxf(mrow[r], mx);
            float al = __expf(mrow[r] - mn);
            float p0 = __expf(s0 - mn);
            float p1 = __expf(s1 - mn);
            float rs = p0 + p1;
#pragma unroll
            for (int off = 8; off >= 1; off >>= 1)
                rs += __shfl_xor(rs, off, 16);
            lrow[r] = lrow[r] * al + rs;
            mrow[r] = mn;
#pragma unroll
            for (int vc = 0; vc < VC; ++vc) acc[vc][r] *= al;
            // C element (row m, col n) -> A-fragment (m, k=n) and (m, k=n+16)
            int m     = r + (grp << 3);
            int laneT = m;
            lP[wave * 512 + laneT * 16 + nn]        = f2bf(p0);
            lP[wave * 512 + (laneT + 16) * 16 + nn] = f2bf(p1);
        }
        __syncthreads();

        // ---- acc += P (16x32) @ V (32xINNER) ----
        const unsigned short* lVb = &lV[buf * VC * 512];
        Frag ap;
        ap.u[0] = *reinterpret_cast<const uint4*>(&lP[wave * 512 + lane * 16]);
        ap.u[1] = *reinterpret_cast<const uint4*>(&lP[wave * 512 + lane * 16 + 8]);
#pragma unroll
        for (int vc = 0; vc < VC; ++vc) {
            Frag b;
            b.u[0] = *reinterpret_cast<const uint4*>(&lVb[vc * 512 + lane * 16]);
            b.u[1] = *reinterpret_cast<const uint4*>(&lVb[vc * 512 + lane * 16 + 8]);
            acc[vc] = wmma_bf16(ap.v, b.v, acc[vc]);
        }
        __syncthreads();   // readers done before this buffer is restaged
    }

    // ---- epilogue: normalize and store ----
#pragma unroll
    for (int r = 0; r < 8; ++r) {
        size_t row = qrow0 + wave * 16 + (grp << 3) + r;
        float inv  = 1.0f / lrow[r];
#pragma unroll
        for (int vc = 0; vc < VC; ++vc) {
            int col = (vc << 4) + nn;
            Og[row * INNER + col] = acc[vc][r] * inv;
        }
        if (WRITE_DENOM) {
            if (nn == 0) denom[row] = lrow[r] * __expf(mrow[r]);  // raw exp-sum
        }
    }
}

// ===========================================================================
// Branch combine (p=1 normalized denominators) + LayerNorm, 1 wave per row.
// Writes into x at row = batch*TSEQ + seqOffset + s.
// ===========================================================================
__global__ __launch_bounds__(256)
void combine_ln_kernel(const float* __restrict__ P0, const float* __restrict__ P1,
                       const float* __restrict__ den0, const float* __restrict__ den1,
                       const float* __restrict__ gamma, const float* __restrict__ beta,
                       float* __restrict__ X, int seqOffset)
{
    const int wave = threadIdx.x >> 5;
    const int lane = threadIdx.x & 31;
    const size_t rin = (size_t)blockIdx.x * 8 + wave;
    const int batch  = (int)(rin >> 10);
    const int s      = (int)(rin & 1023);
    const size_t rout = (size_t)batch * TSEQ + seqOffset + s;

    float w0 = den0[rin], w1 = den1[rin];
    float t  = fmaxf(fabsf(w0) + fabsf(w1), 1e-12f);
    w0 /= t; w1 /= t;

    float v[8];
#pragma unroll
    for (int h = 0; h < 2; ++h) {
        int col  = (h << 7) + (lane << 2);
        float4 a = *reinterpret_cast<const float4*>(&P0[rin * DMODEL + col]);
        float4 b = *reinterpret_cast<const float4*>(&P1[rin * DMODEL + col]);
        v[h * 4 + 0] = a.x * w0 + b.x * w1;
        v[h * 4 + 1] = a.y * w0 + b.y * w1;
        v[h * 4 + 2] = a.z * w0 + b.z * w1;
        v[h * 4 + 3] = a.w * w0 + b.w * w1;
    }
    float s1 = 0.f, s2 = 0.f;
#pragma unroll
    for (int i = 0; i < 8; ++i) { s1 += v[i]; s2 += v[i] * v[i]; }
#pragma unroll
    for (int off = 16; off >= 1; off >>= 1) {
        s1 += __shfl_xor(s1, off, 32);
        s2 += __shfl_xor(s2, off, 32);
    }
    float mean = s1 * (1.f / DMODEL);
    float var  = s2 * (1.f / DMODEL) - mean * mean;
    float rstd = rsqrtf(var + 1e-5f);
#pragma unroll
    for (int h = 0; h < 2; ++h) {
        int col = (h << 7) + (lane << 2);
        float4 o;
        o.x = (v[h * 4 + 0] - mean) * rstd * gamma[col + 0] + beta[col + 0];
        o.y = (v[h * 4 + 1] - mean) * rstd * gamma[col + 1] + beta[col + 1];
        o.z = (v[h * 4 + 2] - mean) * rstd * gamma[col + 2] + beta[col + 2];
        o.w = (v[h * 4 + 3] - mean) * rstd * gamma[col + 3] + beta[col + 3];
        *reinterpret_cast<float4*>(&X[rout * DMODEL + col]) = o;
    }
}

// LayerNorm over 256 features, 1 wave per row (8 rows / 256-thread block).
__global__ __launch_bounds__(256)
void ln_kernel(const float* __restrict__ Xin, const float* __restrict__ gamma,
               const float* __restrict__ beta, float* __restrict__ Y)
{
    const int wave = threadIdx.x >> 5;
    const int lane = threadIdx.x & 31;
    const size_t row = (size_t)blockIdx.x * 8 + wave;

    float v[8];
#pragma unroll
    for (int h = 0; h < 2; ++h) {
        int col  = (h << 7) + (lane << 2);
        float4 a = *reinterpret_cast<const float4*>(&Xin[row * DMODEL + col]);
        v[h * 4 + 0] = a.x; v[h * 4 + 1] = a.y;
        v[h * 4 + 2] = a.z; v[h * 4 + 3] = a.w;
    }
    float s1 = 0.f, s2 = 0.f;
#pragma unroll
    for (int i = 0; i < 8; ++i) { s1 += v[i]; s2 += v[i] * v[i]; }
#pragma unroll
    for (int off = 16; off >= 1; off >>= 1) {
        s1 += __shfl_xor(s1, off, 32);
        s2 += __shfl_xor(s2, off, 32);
    }
    float mean = s1 * (1.f / DMODEL);
    float var  = s2 * (1.f / DMODEL) - mean * mean;
    float rstd = rsqrtf(var + 1e-5f);
#pragma unroll
    for (int h = 0; h < 2; ++h) {
        int col = (h << 7) + (lane << 2);
        float4 o;
        o.x = (v[h * 4 + 0] - mean) * rstd * gamma[col + 0] + beta[col + 0];
        o.y = (v[h * 4 + 1] - mean) * rstd * gamma[col + 1] + beta[col + 1];
        o.z = (v[h * 4 + 2] - mean) * rstd * gamma[col + 2] + beta[col + 2];
        o.w = (v[h * 4 + 3] - mean) * rstd * gamma[col + 3] + beta[col + 3];
        *reinterpret_cast<float4*>(&Y[row * DMODEL + col]) = o;
    }
}

// Final head: out[row] = H[row,:512] . w2 + b2, 1 wave per row.
__global__ __launch_bounds__(256)
void head_kernel(const float* __restrict__ H, const float* __restrict__ w2,
                 const float* __restrict__ b2, float* __restrict__ out)
{
    const int wave = threadIdx.x >> 5;
    const int lane = threadIdx.x & 31;
    const size_t row = (size_t)blockIdx.x * 8 + wave;
    float s = 0.f;
#pragma unroll
    for (int j = 0; j < 16; ++j) {
        int col = j * 32 + lane;
        s += H[row * 512 + col] * w2[col];
    }
#pragma unroll
    for (int off = 16; off >= 1; off >>= 1) s += __shfl_xor(s, off, 32);
    if (lane == 0) out[row] = s + b2[0];
}

// ===========================================================================
// Host orchestration.
// Param leaves assumed flattened in dict-insertion order after the 4 arrays:
//   cross{00,01,10,11}: q.w q.b k.w k.b v.w v.b o.w o.b   -> idx 4..35
//   norms[0].g .b norms[1].g .b                           -> 36..39
//   sa: q.w q.b k.w k.b v.w v.b o.w o.b ln1.g ln1.b
//       ff1.w ff1.b ff2.w ff2.b ln2.g ln2.b               -> 40..55
//   head: l1.w l1.b l2.w l2.b                             -> 56..59
// ===========================================================================
extern "C" void kernel_launch(void* const* d_in, const int* in_sizes, int n_in,
                              void* d_out, int out_size, void* d_ws, size_t ws_size,
                              hipStream_t stream)
{
    (void)in_sizes; (void)n_in; (void)out_size; (void)ws_size;
    const float* words[2]  = { (const float*)d_in[0], (const float*)d_in[1] };
    const float* crossf[2] = { (const float*)d_in[2], (const float*)d_in[3] };
    auto PRM = [&](int i) { return (const float*)d_in[i]; };

    char* base = (char*)d_ws;
    size_t off = 0;
    auto alloc = [&](size_t bytes) -> void* {
        off = (off + 255) & ~(size_t)255;
        void* p = base + off;
        off += bytes;
        return p;
    };

    const int innerv[4] = {128, 64, 64, 64};
    const int dinq[4]   = {144, 144, 64, 64};
    const int dink[4]   = {128, 64, 128, 64};
    const int asel[4]   = {0, 0, 1, 1};
    const int bsel[4]   = {0, 1, 0, 1};

    // persistent concatenated sequence buffer
    float* X = (float*)alloc((size_t)BTROWS * DMODEL * sizeof(float));
    const size_t mark = off;

    unsigned short *Qb[4], *Kb[4], *Vb[4];
    float *Oat[4], *Den[4], *Pp[4];
    for (int p = 0; p < 4; ++p) {
        size_t n = (size_t)BSROWS * innerv[p];
        Qb[p]  = (unsigned short*)alloc(n * 2);
        Kb[p]  = (unsigned short*)alloc(n * 2);
        Vb[p]  = (unsigned short*)alloc(n * 2);   // transposed layout [b][d][t]
        Oat[p] = (float*)alloc(n * 4);
        Den[p] = (float*)alloc((size_t)BSROWS * 4);
        Pp[p]  = (float*)alloc((size_t)BSROWS * DMODEL * 4);
    }

    dim3 blk128(128);
    // ---- 4 cross-attention branches ----
    for (int p = 0; p < 4; ++p) {
        const int pb    = 4 + p * 8;
        const int inner = innerv[p];
        dim3 gproj(inner / 64, BSROWS / 64);
        gemm_kernel<false, false, false, true, false><<<gproj, blk128, 0, stream>>>(
            words[asel[p]], PRM(pb + 0), PRM(pb + 1), nullptr, nullptr, Qb[p],
            nullptr, 0, BSROWS, dinq[p], inner);
        gemm_kernel<false, false, false, true, false><<<gproj, blk128, 0, stream>>>(
            crossf[bsel[p]], PRM(pb + 2), PRM(pb + 3), nullptr, nullptr, Kb[p],
            nullptr, 0, BSROWS, dink[p], inner);
        // V projection written TRANSPOSED so flash can async-stage B-fragments.
        gemm_kernel<false, false, false, false, true><<<gproj, blk128, 0, stream>>>(
            crossf[bsel[p]], PRM(pb + 4), PRM(pb + 5), nullptr, nullptr, nullptr,
            Vb[p], SEQ, BSROWS, dink[p], inner);

        float scale = 1.0f / sqrtf((float)inner);
        if (inner == 128) {
            flash_kernel<128, 4, true, true><<<dim3(SEQ / 64, BATCH), dim3(128), 0, stream>>>(
                Qb[p], Kb[p], Vb[p], Oat[p], Den[p], SEQ, SEQ, scale);
        } else {
            flash_kernel<64, 4, true, true><<<dim3(SEQ / 64, BATCH), dim3(128), 0, stream>>>(
                Qb[p], Kb[p], Vb[p], Oat[p], Den[p], SEQ, SEQ, scale);
        }
        gemm_kernel<false, false, true, false, false><<<dim3(DMODEL / 64, BSROWS / 64),
                                                        blk128, 0, stream>>>(
            Oat[p], PRM(pb + 6), PRM(pb + 7), nullptr, Pp[p], nullptr,
            nullptr, 0, BSROWS, inner, DMODEL);
    }

    // ---- combine branches + per-group LN into x ----
    combine_ln_kernel<<<BSROWS / 8, 256, 0, stream>>>(
        Pp[0], Pp[1], Den[0], Den[1], PRM(36), PRM(37), X, 0);
    combine_ln_kernel<<<BSROWS / 8, 256, 0, stream>>>(
        Pp[2], Pp[3], Den[2], Den[3], PRM(38), PRM(39), X, SEQ);

    // ---- encoder stage (reuse pair arena) ----
    off = mark;
    unsigned short* QS = (unsigned short*)alloc((size_t)BTROWS * DMODEL * 2);
    unsigned short* KS = (unsigned short*)alloc((size_t)BTROWS * DMODEL * 2);
    unsigned short* VS = (unsigned short*)alloc((size_t)BTROWS * DMODEL * 2);
    float* AO  = (float*)alloc((size_t)BTROWS * DMODEL * 4);
    float* X1P = (float*)alloc((size_t)BTROWS * DMODEL * 4);
    float* X1  = (float*)alloc((size_t)BTROWS * DMODEL * 4);
    float* F1  = (float*)alloc((size_t)BTROWS * DMODEL * 4);
    float* X2P = (float*)alloc((size_t)BTROWS * DMODEL * 4);
    float* X2  = (float*)alloc((size_t)BTROWS * DMODEL * 4);
    float* H1  = (float*)alloc((size_t)BTROWS * 512 * 4);

    dim3 gsa(DMODEL / 64, BTROWS / 64);
    gemm_kernel<false, false, false, true, false><<<gsa, blk128, 0, stream>>>(
        X, PRM(40), PRM(41), nullptr, nullptr, QS, nullptr, 0, BTROWS, DMODEL, DMODEL);
    gemm_kernel<false, false, false, true, false><<<gsa, blk128, 0, stream>>>(
        X, PRM(42), PRM(43), nullptr, nullptr, KS, nullptr, 0, BTROWS, DMODEL, DMODEL);
    gemm_kernel<false, false, false, false, true><<<gsa, blk128, 0, stream>>>(
        X, PRM(44), PRM(45), nullptr, nullptr, nullptr, VS, TSEQ, BTROWS, DMODEL, DMODEL);

    flash_kernel<256, 2, false, false><<<dim3(TSEQ / 32, BATCH), dim3(64), 0, stream>>>(
        QS, KS, VS, AO, nullptr, TSEQ, TSEQ, 1.0f / 16.0f);

    gemm_kernel<false, true, true, false, false><<<gsa, blk128, 0, stream>>>(
        AO, PRM(46), PRM(47), X, X1P, nullptr, nullptr, 0, BTROWS, DMODEL, DMODEL);
    ln_kernel<<<BTROWS / 8, 256, 0, stream>>>(X1P, PRM(48), PRM(49), X1);

    gemm_kernel<true, false, true, false, false><<<gsa, blk128, 0, stream>>>(
        X1, PRM(50), PRM(51), nullptr, F1, nullptr, nullptr, 0, BTROWS, DMODEL, DMODEL);
    gemm_kernel<false, true, true, false, false><<<gsa, blk128, 0, stream>>>(
        F1, PRM(52), PRM(53), X1, X2P, nullptr, nullptr, 0, BTROWS, DMODEL, DMODEL);
    ln_kernel<<<BTROWS / 8, 256, 0, stream>>>(X2P, PRM(54), PRM(55), X2);

    gemm_kernel<true, false, true, false, false><<<dim3(512 / 64, BTROWS / 64),
                                                   blk128, 0, stream>>>(
        X2, PRM(56), PRM(57), nullptr, H1, nullptr, nullptr, 0, BTROWS, DMODEL, 512);
    head_kernel<<<BTROWS / 8, 256, 0, stream>>>(H1, PRM(58), PRM(59), (float*)d_out);
}
